// NT_Xent_20624432956306
// MI455X (gfx1250) — matrix operand order repaired and checked by reading, not compiled, not run on hardware
//
#include <hip/hip_runtime.h>
#include <hip/hip_bf16.h>

// Problem constants (match reference).
#define NB   4096            // N
#define DD   64              // projection dim
#define MM   (2 * NB)        // 2N = 8192 rows of sim
#define NCLS 10
#define CSTAGE 64            // columns per LDS panel (8 KB), double buffered
#define NSTAGE (MM / CSTAGE) // 128
// exp(x / TEMP) = exp2(x * (1/TEMP) * log2(e)),  1/TEMP = 2
#define EXP2_SCALE 2.885390081777927f

typedef _Float16 v16h __attribute__((ext_vector_type(16)));
typedef float    v8f  __attribute__((ext_vector_type(8)));

// ---------------------------------------------------------------------------
// CDNA5 async memory->LDS panel copy (no VGPR roundtrip, ASYNCcnt-tracked).
// 32 B per lane: two GLOBAL_LOAD_ASYNC_TO_LDS_B128; offset:4096 advances both
// the LDS and global addresses (CDNA5 ISA).
// ---------------------------------------------------------------------------
__device__ __forceinline__ void async_panel_copy(unsigned lds_off,
                                                 const _Float16* g) {
  asm volatile(
      "global_load_async_to_lds_b128 %0, %1, off\n\t"
      "global_load_async_to_lds_b128 %0, %1, off offset:4096"
      :: "v"(lds_off), "v"(g) : "memory");
}
__device__ __forceinline__ void wait_async_zero() {
  asm volatile("s_wait_asynccnt 0x0" ::: "memory");
}

// ---------------------------------------------------------------------------
// Kernel A: row-normalize p = concat(z_i, z_j); emit f32 (for exact pos) and
// f16 (WMMA operand) copies. One wave per row (lane covers k and k+32).
// ---------------------------------------------------------------------------
__global__ void __launch_bounds__(256)
ntxent_normalize(const float* __restrict__ z_i, const float* __restrict__ z_j,
                 float* __restrict__ pn_f32, _Float16* __restrict__ pn_h) {
  const int wave = threadIdx.x >> 5;
  const int lane = threadIdx.x & 31;
  const int row  = blockIdx.x * 8 + wave;
  const float* src = (row < NB) ? (z_i + (size_t)row * DD)
                                : (z_j + (size_t)(row - NB) * DD);
  float x0 = src[lane];
  float x1 = src[lane + 32];
  float ss = x0 * x0 + x1 * x1;
#pragma unroll
  for (int m = 16; m >= 1; m >>= 1) ss += __shfl_xor(ss, m, 32);
  const float inv = 1.0f / fmaxf(sqrtf(ss), 1e-8f);
  const float y0 = x0 * inv, y1 = x1 * inv;
  pn_f32[(size_t)row * DD + lane]      = y0;
  pn_f32[(size_t)row * DD + lane + 32] = y1;
  pn_h  [(size_t)row * DD + lane]      = (_Float16)y0;
  pn_h  [(size_t)row * DD + lane + 32] = (_Float16)y1;
}

// ---------------------------------------------------------------------------
// Kernel B: fused sim + softmax-denominator. Each wave owns a 16-row strip.
// Double-buffered 64-column LDS panels filled by async-to-LDS DMA overlap the
// next panel's fill with this panel's 8 WMMAs + exps. One s_wait_asynccnt +
// one barrier per stage. Full row sum (diag included) is produced branch-free
// with EXEC all-ones; the diagonal term is subtracted exactly in kernel C.
// ---------------------------------------------------------------------------
__global__ void __launch_bounds__(256)
ntxent_simsum(const _Float16* __restrict__ pn_h, float* __restrict__ rowSum) {
  __shared__ _Float16 sB[2][CSTAGE * DD];     // two 8 KB panels
  const int wave = threadIdx.x >> 5;
  const int lane = threadIdx.x & 31;
  const int R0   = (blockIdx.x * 8 + wave) * 16;   // first row of this strip

  // LDS byte offsets of this thread's 32 B slot in each panel (flat shared
  // pointer low 32 bits == LDS offset on AMDGCN).
  const unsigned lds0 = (unsigned)(size_t)(&sB[0][0]) + threadIdx.x * 16u;
  const unsigned lds1 = (unsigned)(size_t)(&sB[1][0]) + threadIdx.x * 16u;

  // A fragments (ISA 16-bit A 16x32 layout):
  // lanes 0-15: row = lane,    K in {kc+0..7,  kc+16..23}
  // lanes16-31: row = lane-16, K in {kc+8..15, kc+24..31}
  const int mrow = R0 + (lane & 15);
  const int kbA  = (lane < 16) ? 0 : 8;
  v16h a0, a1;
  {
    const _Float16* pr = pn_h + (size_t)mrow * DD;
    ((uint4*)&a0)[0] = *(const uint4*)(pr + kbA);
    ((uint4*)&a0)[1] = *(const uint4*)(pr + kbA + 16);
    ((uint4*)&a1)[0] = *(const uint4*)(pr + 32 + kbA);
    ((uint4*)&a1)[1] = *(const uint4*)(pr + 32 + kbA + 16);
  }

  float acc[8];
#pragma unroll
  for (int v = 0; v < 8; ++v) acc[v] = 0.0f;

  // B fragment layout (16-bit B 32x16): lane n<16 -> col n, K in {kc+0..7, kc+16..23};
  // lane n>=16 -> col n-16, K in {kc+8..15, kc+24..31} (matches A).
  const int kbB = (lane < 16) ? 0 : 8;
  const int ncl = lane & 15;

  // Prologue: start DMA of panel 0.
  async_panel_copy(lds0, pn_h + (size_t)threadIdx.x * 8);

  for (int s = 0; s < NSTAGE; ++s) {
    wait_async_zero();        // this wave's panel-s writes have landed in LDS
    __syncthreads();          // everyone's landed; prior panel reads retired

    if (s + 1 < NSTAGE) {     // kick DMA for the next panel (other buffer)
      async_panel_copy(((s + 1) & 1) ? lds1 : lds0,
                       pn_h + (size_t)(s + 1) * CSTAGE * DD +
                           (size_t)threadIdx.x * 8);
    }

    const _Float16* panel = (s & 1) ? &sB[1][0] : &sB[0][0];
#pragma unroll
    for (int j = 0; j < CSTAGE / 16; ++j) {
      const _Float16* pc = panel + (size_t)(j * 16 + ncl) * DD;
      v16h b0, b1;
      ((uint4*)&b0)[0] = *(const uint4*)(pc + kbB);
      ((uint4*)&b0)[1] = *(const uint4*)(pc + kbB + 16);
      ((uint4*)&b1)[0] = *(const uint4*)(pc + 32 + kbB);
      ((uint4*)&b1)[1] = *(const uint4*)(pc + 32 + kbB + 16);

      v8f c = {};
      c = __builtin_amdgcn_wmma_f32_16x16x32_f16(false, a0, false, b0,
                                                 (short)0, c, false, false);
      c = __builtin_amdgcn_wmma_f32_16x16x32_f16(false, a1, false, b1,
                                                 (short)0, c, false, false);

      // Branch-free hot path: one v_mul + one v_exp per element.
#pragma unroll
      for (int v = 0; v < 8; ++v)
        acc[v] += __builtin_amdgcn_exp2f(c[v] * EXP2_SCALE);
    }
  }

  // Each row's sum lives across the 16 lanes of one half -> xor-reduce.
#pragma unroll
  for (int v = 0; v < 8; ++v) {
#pragma unroll
    for (int m = 8; m >= 1; m >>= 1) acc[v] += __shfl_xor(acc[v], m, 32);
  }
  if (lane == 0) {
#pragma unroll
    for (int v = 0; v < 8; ++v) rowSum[R0 + v] = acc[v];
  }
  if (lane == 16) {
#pragma unroll
    for (int v = 0; v < 8; ++v) rowSum[R0 + 8 + v] = acc[v];
  }
}

// ---------------------------------------------------------------------------
// Kernel C: per-row pos (exact f32 dot), diagonal removal, loss1/loss2.
// The diagonal term of rowSum went through the f16 WMMA path, so it is
// reproduced from the f16-rounded values: exp(2 * sum h(x_k)^2).
// ---------------------------------------------------------------------------
__global__ void __launch_bounds__(256)
ntxent_rows(const float* __restrict__ pn_f32, const float* __restrict__ rowSum,
            const float* __restrict__ probs, const int* __restrict__ target,
            float* __restrict__ c1, float* __restrict__ c2) {
  const int r = blockIdx.x * blockDim.x + threadIdx.x;
  if (r >= MM) return;
  const int q = (r < NB) ? r + NB : r - NB;              // (r + N) % 2N
  const float* pr = pn_f32 + (size_t)r * DD;
  const float* pq = pn_f32 + (size_t)q * DD;
  float dot = 0.0f, sd = 0.0f;
#pragma unroll
  for (int k = 0; k < DD; ++k) {
    dot = fmaf(pr[k], pq[k], dot);
    float h = (float)(_Float16)pr[k];                    // f16-rounded value
    sd = fmaf(h, h, sd);
  }
  const float pos  = dot * 2.0f;                         // * 1/TEMP
  const float diag = __builtin_amdgcn_exp2f(sd * EXP2_SCALE);
  c1[r] = __logf(rowSum[r] - diag) - pos;                // lse(row\{r}) - pos

  const int t = target[(r < NB) ? r : r - NB];
  float s = __expf(pos);                                 // values bounded: no
  const float* pb = probs + (size_t)r * NCLS;            // stabilization needed
#pragma unroll
  for (int cidx = 0; cidx < NCLS; ++cidx)
    s += (cidx == t) ? 0.0f : __expf(pb[cidx]);
  c2[r] = __logf(s) - pos;
}

// ---------------------------------------------------------------------------
// Kernel D: deterministic single-block reduction -> (loss1, loss2) / 2N.
// ---------------------------------------------------------------------------
__global__ void __launch_bounds__(256)
ntxent_final(const float* __restrict__ c1, const float* __restrict__ c2,
             float* __restrict__ out) {
  __shared__ float s1[256], s2[256];
  float a = 0.0f, b = 0.0f;
  for (int i = threadIdx.x; i < MM; i += 256) { a += c1[i]; b += c2[i]; }
  s1[threadIdx.x] = a;
  s2[threadIdx.x] = b;
  __syncthreads();
  for (int off = 128; off >= 1; off >>= 1) {
    if (threadIdx.x < off) {
      s1[threadIdx.x] += s1[threadIdx.x + off];
      s2[threadIdx.x] += s2[threadIdx.x + off];
    }
    __syncthreads();
  }
  if (threadIdx.x == 0) {
    out[0] = s1[0] / (float)MM;
    out[1] = s2[0] / (float)MM;
  }
}

// ---------------------------------------------------------------------------
extern "C" void kernel_launch(void* const* d_in, const int* in_sizes, int n_in,
                              void* d_out, int out_size, void* d_ws,
                              size_t ws_size, hipStream_t stream) {
  (void)in_sizes; (void)n_in; (void)out_size; (void)ws_size;
  const float* z_i    = (const float*)d_in[0];
  const float* z_j    = (const float*)d_in[1];
  const float* probs  = (const float*)d_in[2];
  const int*   target = (const int*)d_in[3];
  // d_in[4] (neg_idx) encodes the structured mask "all columns except self and
  // positive" — handled in closed form, never read.

  float*    pn_f32 = (float*)d_ws;                          // 2 MB
  _Float16* pn_h   = (_Float16*)(pn_f32 + (size_t)MM * DD); // 1 MB
  float*    rowSum = (float*)(pn_h + (size_t)MM * DD);      // 32 KB
  float*    c1     = rowSum + MM;                           // 32 KB
  float*    c2     = c1 + MM;                               // 32 KB

  ntxent_normalize<<<MM / 8, 256, 0, stream>>>(z_i, z_j, pn_f32, pn_h);
  ntxent_simsum<<<(MM / 16) / 8, 256, 0, stream>>>(pn_h, rowSum);
  ntxent_rows<<<MM / 256, 256, 0, stream>>>(pn_f32, rowSum, probs, target,
                                            c1, c2);
  ntxent_final<<<1, 256, 0, stream>>>(c1, c2, (float*)d_out);
}
